// RNN_downscaler_12790412607808
// MI455X (gfx1250) — matrix operand order repaired and checked by reading, not compiled, not run on hardware
//
#include <hip/hip_runtime.h>
#include <hip/hip_bf16.h>
#include <math.h>

// ---------------------------------------------------------------------------
// Types for CDNA5 WMMA (wave32, 16x16x32 bf16 -> f32)
// ---------------------------------------------------------------------------
typedef __bf16 bf16_t;
typedef __attribute__((ext_vector_type(16))) __bf16 v16bf;
typedef __attribute__((ext_vector_type(8)))  __bf16 v8bf;
typedef __attribute__((ext_vector_type(8)))  float  v8f;
typedef unsigned int  u32x4 __attribute__((ext_vector_type(4)));
typedef int           i32x4 __attribute__((ext_vector_type(4)));
typedef int           i32x8 __attribute__((ext_vector_type(8)));

union V16U { v16bf v; v8bf h[2]; };

// Problem constants
#define BB   16
#define TT   128
#define LL   64
#define DD   256
#define HH   24
#define NHD  8
#define DH   32
#define NSEQ 2048          // B*T
#define NROW 131072        // B*T*L

#if defined(__HIP_DEVICE_COMPILE__) && __has_builtin(__builtin_amdgcn_tensor_load_to_lds)
#define USE_TDM 1
#else
#define USE_TDM 0
#endif

// --- device-pass diagnostics (visible in compile stderr; host pass silent) ---
#if defined(__HIP_DEVICE_COMPILE__)
#if USE_TDM
#warning "DEVICE PROBE: tensor_load_to_lds builtin AVAILABLE (TDM path compiled)"
#else
#warning "DEVICE PROBE: tensor_load_to_lds builtin NOT available (fallback staging)"
#endif
#if defined(__clang_major__) && __clang_major__ >= 23
#warning "DEVICE PROBE: clang >= 23 (6-arg tensor_load_to_lds form)"
#else
#warning "DEVICE PROBE: clang < 23 (5-arg tensor_load_to_lds form)"
#endif
#endif

// ---------------------------------------------------------------------------
// TDM: async 2D tile load Global -> LDS (Tensor Data Mover, gfx1250)
// Descriptor layout per CDNA5 ISA ch.8 (D# groups 0/1).
// tile: tile1 rows x tile0 elements (bf16, 2B), row stride = strideElems.
// ---------------------------------------------------------------------------
#if USE_TDM
static __device__ inline void tdm_load_tile_2d(unsigned lds_off, const void* gptr,
                                               unsigned long long strideElems,
                                               int tile0, int tile1)
{
    unsigned long long ga = (unsigned long long)(size_t)gptr;
    const unsigned DIM = 1u << 30;   // generous tensor dims: tiles are in-bounds
    u32x4 g0;
    g0[0] = 1u;                                   // count=1 valid descriptor
    g0[1] = lds_off;                              // LDS byte address
    g0[2] = (unsigned)ga;                         // global_addr[31:0]
    g0[3] = (unsigned)(ga >> 32) | (2u << 30);    // global_addr[56:32] | type=2
    i32x8 g1;
    g1[0] = (int)(1u << 16);                      // data_size=1 (2 bytes)
    g1[1] = (int)((DIM & 0xFFFFu) << 16);         // tensor_dim0[15:0]
    g1[2] = (int)(((DIM >> 16) & 0xFFFFu) | ((DIM & 0xFFFFu) << 16));   // dim0 hi | dim1 lo
    g1[3] = (int)(((DIM >> 16) & 0xFFFFu) | ((unsigned)tile0 << 16));   // dim1 hi | tile_dim0
    g1[4] = (int)((unsigned)tile1);               // tile_dim1 | tile_dim2=0
    g1[5] = (int)(unsigned)strideElems;           // tensor_dim0_stride[31:0]
    g1[6] = (int)((unsigned)(strideElems >> 32) & 0xFFFFu);  // stride0[47:32] | stride1 lo=0
    g1[7] = 0;
    i32x4 z4 = {0, 0, 0, 0};
#if defined(__clang_major__) && __clang_major__ >= 23
    i32x8 z8 = {0, 0, 0, 0, 0, 0, 0, 0};
    __builtin_amdgcn_tensor_load_to_lds(g0, g1, z4, z4, z8, 0);
#else
    __builtin_amdgcn_tensor_load_to_lds(g0, g1, z4, z4, 0);
#endif
}

static __device__ inline void tdm_wait0()
{
#if __has_builtin(__builtin_amdgcn_s_wait_tensorcnt)
    __builtin_amdgcn_s_wait_tensorcnt(0);
#else
    asm volatile("s_wait_tensorcnt 0" ::: "memory");
#endif
}
#endif

// ---------------------------------------------------------------------------
// 1. Sequential RNN scan: h_t = relu(x0_t*W_ih + b_ih + h_{t-1}@W_hh.T + b_hh)
// ---------------------------------------------------------------------------
__global__ void rnn_kernel(const float* __restrict__ x_cont,
                           const float* __restrict__ z0,
                           const float* __restrict__ W_ih,
                           const float* __restrict__ W_hh,
                           const float* __restrict__ b_ih,
                           const float* __restrict__ b_hh,
                           float* __restrict__ h_out)
{
    __shared__ float sW[LL * LL];
    __shared__ float sH[BB * LL];
    int tid = threadIdx.x;               // 0..1023
    for (int i = tid; i < LL * LL; i += 1024) sW[i] = W_hh[i];
    int b = tid >> 6;
    int l = tid & 63;
    sH[tid] = z0[l];
    __syncthreads();
    float wih = W_ih[l];
    float bias = b_ih[l] + b_hh[l];
    for (int t = 0; t < TT; ++t) {
        float x0 = x_cont[(size_t)(b * TT + t) * 97];
        float acc = x0 * wih + bias;
        const float* hb = &sH[b * LL];
        const float* wl = &sW[l * LL];
        #pragma unroll 8
        for (int j = 0; j < LL; ++j) acc += hb[j] * wl[j];
        float hv = fmaxf(acc, 0.0f);
        h_out[(size_t)(b * TT + t) * LL + l] = hv;
        __syncthreads();
        sH[tid] = hv;
        __syncthreads();
    }
}

// ---------------------------------------------------------------------------
// 2. z = h + fourier_h @ Wfp.T + bfp
// ---------------------------------------------------------------------------
__global__ void z_kernel(const float* __restrict__ x_cont,
                         const float* __restrict__ h,
                         const float* __restrict__ Uf,
                         const float* __restrict__ Wfp,
                         const float* __restrict__ bfp,
                         float* __restrict__ z)
{
    __shared__ float sF[HH];
    int n = blockIdx.x;
    int l = threadIdx.x;     // 0..63
    if (l < HH) {
        float s = 0.0f;
        #pragma unroll
        for (int f = 0; f < 4; ++f)
            s += x_cont[(size_t)n * 97 + 1 + l * 4 + f] * Uf[l * 4 + f];
        sF[l] = s;
    }
    __syncthreads();
    float acc = h[(size_t)n * LL + l] + bfp[l];
    #pragma unroll
    for (int hh = 0; hh < HH; ++hh) acc += sF[hh] * Wfp[l * HH + hh];
    z[(size_t)n * LL + l] = acc;
}

// ---------------------------------------------------------------------------
// 3. FE_qkv = feat_embed @ W_in.T  (64 x 768), rank-1 collapse of tokens@W_in.T
// ---------------------------------------------------------------------------
__global__ void feqkv_kernel(const float* __restrict__ feat_embed,
                             const float* __restrict__ W_in,
                             float* __restrict__ FE)
{
    int idx = blockIdx.x * 256 + threadIdx.x;   // < 64*768
    int l = idx / 768;
    int o = idx % 768;
    const float* fe = feat_embed + (size_t)l * DD;
    const float* wr = W_in + (size_t)o * DD;
    float s = 0.0f;
    #pragma unroll 8
    for (int d = 0; d < DD; ++d) s += fe[d] * wr[d];
    FE[idx] = s;
}

// ---------------------------------------------------------------------------
// 4. f32 -> bf16 weight conversion
// ---------------------------------------------------------------------------
__global__ void cvt_kernel(const float* __restrict__ in, bf16_t* __restrict__ out, int n)
{
    for (int i = blockIdx.x * blockDim.x + threadIdx.x; i < n; i += gridDim.x * blockDim.x)
        out[i] = (bf16_t)in[i];
}

// ---------------------------------------------------------------------------
// 5. Attention: one block per sequence n; wave w == head w; WMMA both matmuls.
// ---------------------------------------------------------------------------
__global__ void attn_kernel(const float* __restrict__ z,
                            const float* __restrict__ FE,
                            const float* __restrict__ b_in,
                            bf16_t* __restrict__ ao)
{
    extern __shared__ char smem[];
    bf16_t* sQ   = (bf16_t*)smem;
    bf16_t* sK   = sQ + LL * DD;
    bf16_t* sV   = sK + LL * DD;
    bf16_t* sAtt = sV + LL * DD;            // 8 waves * 16*64

    int n   = blockIdx.x;
    int tid = threadIdx.x;

    for (int idx = tid; idx < LL * 768; idx += 256) {
        int l = idx / 768;
        int o = idx % 768;
        float val = z[(size_t)n * LL + l] * FE[idx] + b_in[o];
        bf16_t bv = (bf16_t)val;
        if (o < DD)            sQ[l * DD + o]        = bv;
        else if (o < 2 * DD)   sK[l * DD + (o - DD)] = bv;
        else                   sV[l * DD + (o - 2 * DD)] = bv;
    }
    __syncthreads();

    int w    = tid >> 5;        // head index
    int lane = tid & 31;
    int c    = lane & 15;
    int half = lane >> 4;
    bf16_t* sAw = sAtt + w * 16 * LL;
    const float scale = 0.17677669529663687f;   // 1/sqrt(32)
    const v8f zero8 = {0, 0, 0, 0, 0, 0, 0, 0};

    for (int mb = 0; mb < 4; ++mb) {
        const bf16_t* ap = sQ + (size_t)(mb * 16 + c) * DD + w * DH + half * 16;
        V16U ua;
        ua.h[0] = *(const v8bf*)(ap);
        ua.h[1] = *(const v8bf*)(ap + 8);
        v8f acc[4];
        #pragma unroll
        for (int nb = 0; nb < 4; ++nb) {
            const bf16_t* bp = sK + (size_t)(nb * 16 + c) * DD + w * DH + 16 * half;
            v16bf Bv = *(const v16bf*)bp;
            acc[nb] = __builtin_amdgcn_wmma_f32_16x16x32_bf16(
                false, ua.v, false, Bv, (short)0, zero8, false, false);
        }
        #pragma unroll
        for (int vr = 0; vr < 8; ++vr) {
            float sv[4];
            float mx = -3.0e38f;
            #pragma unroll
            for (int nb = 0; nb < 4; ++nb) {
                sv[nb] = acc[nb][vr] * scale;
                mx = fmaxf(mx, sv[nb]);
            }
            for (int off = 1; off < 16; off <<= 1)
                mx = fmaxf(mx, __shfl_xor(mx, off, 32));
            float ssum = 0.0f;
            #pragma unroll
            for (int nb = 0; nb < 4; ++nb) { sv[nb] = __expf(sv[nb] - mx); ssum += sv[nb]; }
            for (int off = 1; off < 16; off <<= 1)
                ssum += __shfl_xor(ssum, off, 32);
            float inv = 1.0f / ssum;
            int rl = vr + 8 * half;
            #pragma unroll
            for (int nb = 0; nb < 4; ++nb)
                sAw[rl * LL + nb * 16 + c] = (bf16_t)(sv[nb] * inv);
        }
        asm volatile("s_wait_dscnt 0" ::: "memory");

        #pragma unroll
        for (int nbv = 0; nbv < 2; ++nbv) {
            v8f acc2 = zero8;
            #pragma unroll
            for (int k0 = 0; k0 < LL; k0 += 32) {
                const bf16_t* ap2 = sAw + (size_t)c * LL + k0 + half * 16;
                V16U ua2;
                ua2.h[0] = *(const v8bf*)(ap2);
                ua2.h[1] = *(const v8bf*)(ap2 + 8);
                v16bf Bv2;
                #pragma unroll
                for (int i = 0; i < 16; ++i)
                    Bv2[i] = sV[(size_t)(k0 + 16 * half + i) * DD + w * DH + nbv * 16 + c];
                acc2 = __builtin_amdgcn_wmma_f32_16x16x32_bf16(
                    false, ua2.v, false, Bv2, (short)0, acc2, false, false);
            }
            #pragma unroll
            for (int vr = 0; vr < 8; ++vr) {
                int m = mb * 16 + vr + 8 * half;
                ao[((size_t)n * LL + m) * DD + w * DH + nbv * 16 + c] = (bf16_t)acc2[vr];
            }
        }
    }
}

// ---------------------------------------------------------------------------
// 6. bf16 WMMA GEMM v2: Y[m,n] = sum_k X[m,k] * W[n,k]
//    Block tile 128(M) x 64(N); wave tile 32x32 (2x2 WMMA tiles).
//    X chunks (128 x 64 bf16) staged into double-buffered LDS by the TDM
//    (tensor_load_to_lds + s_wait_tensorcnt), overlapping DMA with WMMA.
//    MODE 0: Yf = acc (f32).  MODE 1: Yh = bf16(relu(acc + bias[n])).
// ---------------------------------------------------------------------------
template <int MODE>
__global__ void gemm_bf16_kernel(const bf16_t* __restrict__ X,
                                 const bf16_t* __restrict__ W,
                                 const float* __restrict__ bias,
                                 float* __restrict__ Yf,
                                 bf16_t* __restrict__ Yh,
                                 int M, int N, int K)
{
    __shared__ bf16_t sX[2][128 * 64];   // 2 x 16 KB double buffer

    const int bn = blockIdx.x;           // N / 64
    const int bm = blockIdx.y;           // M / 128
    const int tid  = threadIdx.x;
    const int wv   = tid >> 5;
    const int lane = tid & 31;
    const int c    = lane & 15;
    const int half = lane >> 4;
    const int mq   = wv >> 1;            // 0..3 : 32-row strip within block
    const int nq   = wv & 1;             // 0..1 : 32-col strip within block

    v8f acc[2][2];
    const v8f zero8 = {0, 0, 0, 0, 0, 0, 0, 0};
    #pragma unroll
    for (int mb = 0; mb < 2; ++mb)
        #pragma unroll
        for (int nb = 0; nb < 2; ++nb) acc[mb][nb] = zero8;

    const bf16_t* Xblk = X + (size_t)bm * 128 * K;

#if USE_TDM
    if (wv == 0)
        tdm_load_tile_2d((unsigned)(size_t)&sX[0][0], Xblk,
                         (unsigned long long)K, 64, 128);
#endif

    const int nchunks = K >> 6;
    for (int ch = 0; ch < nchunks; ++ch) {
        const int cur = ch & 1;
        const int kc  = ch << 6;
#if USE_TDM
        if (wv == 0) tdm_wait0();
        __syncthreads();
        if (wv == 0 && ch + 1 < nchunks)
            tdm_load_tile_2d((unsigned)(size_t)&sX[cur ^ 1][0],
                             Xblk + (size_t)(kc + 64),
                             (unsigned long long)K, 64, 128);
#else
        {   // cooperative staging fallback: 256 threads x 32 bf16
            const int r = tid >> 1;
            const int ch32 = (tid & 1) * 32;
            const bf16_t* src = Xblk + (size_t)r * K + kc + ch32;
            bf16_t* dst = &sX[cur][r * 64 + ch32];
            *(v16bf*)dst = *(const v16bf*)src;
            *(v16bf*)(dst + 16) = *(const v16bf*)(src + 16);
        }
        __syncthreads();
#endif
        const bf16_t* sXc = &sX[cur][0];
        #pragma unroll
        for (int ks = 0; ks < 2; ++ks) {
            const int kl = ks * 32;
            V16U a[2];
            #pragma unroll
            for (int mb = 0; mb < 2; ++mb) {
                const bf16_t* ap = sXc + (mq * 32 + mb * 16 + c) * 64 + kl + half * 16;
                a[mb].h[0] = *(const v8bf*)(ap);
                a[mb].h[1] = *(const v8bf*)(ap + 8);
            }
            #pragma unroll
            for (int nb = 0; nb < 2; ++nb) {
                const bf16_t* bp = W + (size_t)(bn * 64 + nq * 32 + nb * 16 + c) * K
                                     + kc + kl + 16 * half;
                v16bf Bv = *(const v16bf*)bp;
                #pragma unroll
                for (int mb = 0; mb < 2; ++mb)
                    acc[mb][nb] = __builtin_amdgcn_wmma_f32_16x16x32_bf16(
                        false, a[mb].v, false, Bv, (short)0, acc[mb][nb], false, false);
            }
        }
        __syncthreads();
    }

    #pragma unroll
    for (int nb = 0; nb < 2; ++nb) {
        const int nc = bn * 64 + nq * 32 + nb * 16 + c;
        const float bv = (MODE == 1 && bias) ? bias[nc] : 0.0f;
        #pragma unroll
        for (int mb = 0; mb < 2; ++mb) {
            #pragma unroll
            for (int vr = 0; vr < 8; ++vr) {
                size_t m = (size_t)bm * 128 + mq * 32 + mb * 16 + vr + 8 * half;
                float v = acc[mb][nb][vr];
                if (MODE == 1) {
                    v = fmaxf(v + bv, 0.0f);
                    Yh[m * (size_t)N + nc] = (bf16_t)v;
                } else {
                    Yf[m * (size_t)N + nc] = v;
                }
            }
        }
    }
}

// ---------------------------------------------------------------------------
// 7. LN1: t1 = LN(tokens + ao_proj + b_ao) * g1 + be1  (wave-per-row, D=256)
// ---------------------------------------------------------------------------
__global__ void ln1_kernel(const float* __restrict__ Y1,
                           const float* __restrict__ z,
                           const float* __restrict__ feat_embed,
                           const float* __restrict__ b_ao,
                           const float* __restrict__ g1,
                           const float* __restrict__ be1,
                           float* __restrict__ t1f,
                           bf16_t* __restrict__ t1h)
{
    int w    = threadIdx.x >> 5;
    int lane = threadIdx.x & 31;
    size_t row = (size_t)blockIdx.x * 8 + w;
    int l = (int)(row & 63);
    float zr = z[row];
    float x[8];
    float s1 = 0.0f, s2 = 0.0f;
    #pragma unroll
    for (int j = 0; j < 8; ++j) {
        int d = j * 32 + lane;
        float v = zr * feat_embed[(size_t)l * DD + d] + Y1[row * DD + d] + b_ao[d];
        x[j] = v;
        s1 += v;
        s2 += v * v;
    }
    for (int off = 1; off < 32; off <<= 1) {
        s1 += __shfl_xor(s1, off, 32);
        s2 += __shfl_xor(s2, off, 32);
    }
    float mu = s1 * (1.0f / DD);
    float var = s2 * (1.0f / DD) - mu * mu;
    float rstd = rsqrtf(var + 1e-5f);
    #pragma unroll
    for (int j = 0; j < 8; ++j) {
        int d = j * 32 + lane;
        float t = (x[j] - mu) * rstd * g1[d] + be1[d];
        t1f[row * DD + d] = t;
        t1h[row * DD + d] = (bf16_t)t;
    }
}

// ---------------------------------------------------------------------------
// 8. LN2 + delta: t2 = LN(t1 + ffn + b2)*g2 + be2; delta[row] = t2·Wop + bop
// ---------------------------------------------------------------------------
__global__ void ln2_delta_kernel(const float* __restrict__ t1f,
                                 const float* __restrict__ F,
                                 const float* __restrict__ b2,
                                 const float* __restrict__ g2,
                                 const float* __restrict__ be2,
                                 const float* __restrict__ Wop,
                                 const float* __restrict__ bop,
                                 float* __restrict__ delta)
{
    int w    = threadIdx.x >> 5;
    int lane = threadIdx.x & 31;
    size_t row = (size_t)blockIdx.x * 8 + w;
    float x[8];
    float s1 = 0.0f, s2 = 0.0f;
    #pragma unroll
    for (int j = 0; j < 8; ++j) {
        int d = j * 32 + lane;
        float v = t1f[row * DD + d] + F[row * DD + d] + b2[d];
        x[j] = v;
        s1 += v;
        s2 += v * v;
    }
    for (int off = 1; off < 32; off <<= 1) {
        s1 += __shfl_xor(s1, off, 32);
        s2 += __shfl_xor(s2, off, 32);
    }
    float mu = s1 * (1.0f / DD);
    float var = s2 * (1.0f / DD) - mu * mu;
    float rstd = rsqrtf(var + 1e-5f);
    float dl = 0.0f;
    #pragma unroll
    for (int j = 0; j < 8; ++j) {
        int d = j * 32 + lane;
        float t = (x[j] - mu) * rstd * g2[d] + be2[d];
        dl += t * Wop[d];
    }
    for (int off = 1; off < 32; off <<= 1)
        dl += __shfl_xor(dl, off, 32);
    if (lane == 0) delta[row] = dl + bop[0];
}

// ---------------------------------------------------------------------------
// 9. gate/sigmoid, z_out, output projection o = z_out @ A.T + c
// ---------------------------------------------------------------------------
__global__ void final_kernel(const float* __restrict__ z,
                             const float* __restrict__ delta,
                             const float* __restrict__ Wg,
                             const float* __restrict__ bg,
                             const float* __restrict__ A,
                             const float* __restrict__ cvec,
                             float* __restrict__ out_o,
                             float* __restrict__ out_z)
{
    __shared__ float sz[LL];
    __shared__ float szo[LL];
    int n = blockIdx.x;
    int l = threadIdx.x;
    sz[l] = z[(size_t)n * LL + l];
    __syncthreads();
    float g = bg[l];
    #pragma unroll 8
    for (int j = 0; j < LL; ++j) g += sz[j] * Wg[l * LL + j];
    g = 1.0f / (1.0f + __expf(-g));
    float zo = sz[l] + g * delta[(size_t)n * LL + l];
    out_z[(size_t)n * LL + l] = zo;
    szo[l] = zo;
    __syncthreads();
    if (l < HH) {
        float o = cvec[l];
        #pragma unroll 8
        for (int j = 0; j < LL; ++j) o += szo[j] * A[l * LL + j];
        out_o[(size_t)n * HH + l] = o;
    }
}

// ---------------------------------------------------------------------------
// Launch
// ---------------------------------------------------------------------------
extern "C" void kernel_launch(void* const* d_in, const int* in_sizes, int n_in,
                              void* d_out, int out_size, void* d_ws, size_t ws_size,
                              hipStream_t stream)
{
    const float* x_cont     = (const float*)d_in[0];
    const float* z0         = (const float*)d_in[1];
    const float* W_ih       = (const float*)d_in[2];
    const float* W_hh       = (const float*)d_in[3];
    const float* b_ih       = (const float*)d_in[4];
    const float* b_hh       = (const float*)d_in[5];
    const float* Uf         = (const float*)d_in[6];
    const float* Wfp        = (const float*)d_in[7];
    const float* bfp        = (const float*)d_in[8];
    const float* feat_embed = (const float*)d_in[9];
    const float* W_in       = (const float*)d_in[10];
    const float* b_in       = (const float*)d_in[11];
    const float* W_ao       = (const float*)d_in[12];
    const float* b_ao       = (const float*)d_in[13];
    const float* g1         = (const float*)d_in[14];
    const float* be1        = (const float*)d_in[15];
    const float* W1         = (const float*)d_in[16];
    const float* b1         = (const float*)d_in[17];
    const float* W2         = (const float*)d_in[18];
    const float* b2         = (const float*)d_in[19];
    const float* g2         = (const float*)d_in[20];
    const float* be2        = (const float*)d_in[21];
    const float* Wop        = (const float*)d_in[22];
    const float* bop        = (const float*)d_in[23];
    const float* Wg         = (const float*)d_in[24];
    const float* bg         = (const float*)d_in[25];
    const float* Amat       = (const float*)d_in[26];
    const float* cvec       = (const float*)d_in[27];

    // bump-allocate workspace (256B aligned)
    char* p = (char*)d_ws;
    auto alloc = [&](size_t bytes) -> char* {
        char* r = p;
        p += (bytes + 255) & ~(size_t)255;
        return r;
    };
    float*  h     = (float*)alloc((size_t)NROW * 4);
    float*  z     = (float*)alloc((size_t)NROW * 4);
    float*  FE    = (float*)alloc((size_t)LL * 768 * 4);
    bf16_t* WaoH  = (bf16_t*)alloc((size_t)DD * DD * 2);
    bf16_t* W1H   = (bf16_t*)alloc((size_t)4 * DD * DD * 2);
    bf16_t* W2H   = (bf16_t*)alloc((size_t)4 * DD * DD * 2);
    float*  delta = (float*)alloc((size_t)NROW * 4);
    bf16_t* aoH   = (bf16_t*)alloc((size_t)NROW * DD * 2);
    float*  Ybuf  = (float*)alloc((size_t)NROW * DD * 4);   // ao-proj, then FFN out
    float*  t1f   = (float*)alloc((size_t)NROW * DD * 4);
    bf16_t* t1h   = (bf16_t*)alloc((size_t)NROW * DD * 2);
    bf16_t* U     = (bf16_t*)alloc((size_t)NROW * 4 * DD * 2);

    float* out_o = (float*)d_out;                 // (B,T,H) = 49152
    float* out_z = out_o + (size_t)BB * TT * HH;  // (B,T,L) = 131072

    // weight conversions (independent)
    cvt_kernel<<<256, 256, 0, stream>>>(W_ao, WaoH, DD * DD);
    cvt_kernel<<<1024, 256, 0, stream>>>(W1, W1H, 4 * DD * DD);
    cvt_kernel<<<1024, 256, 0, stream>>>(W2, W2H, 4 * DD * DD);

    // sequential RNN + small precomputes
    rnn_kernel<<<1, 1024, 0, stream>>>(x_cont, z0, W_ih, W_hh, b_ih, b_hh, h);
    feqkv_kernel<<<192, 256, 0, stream>>>(feat_embed, W_in, FE);
    z_kernel<<<NSEQ, 64, 0, stream>>>(x_cont, h, Uf, Wfp, bfp, z);

    // attention (WMMA), LDS = 3*64*256 bf16 + 8*16*64 bf16 = 114688 B
    const unsigned attn_smem = (3 * LL * DD + 8 * 16 * LL) * sizeof(bf16_t);
    attn_kernel<<<NSEQ, 256, attn_smem, stream>>>(z, FE, b_in, aoH);

    // ao @ W_ao.T  -> f32        (M=131072, N=256, K=256)
    gemm_bf16_kernel<0><<<dim3(DD / 64, NROW / 128), 256, 0, stream>>>(
        aoH, WaoH, nullptr, Ybuf, nullptr, NROW, DD, DD);
    // LN1 (adds tokens + b_ao, emits f32 residual + bf16 GEMM input)
    ln1_kernel<<<NROW / 8, 256, 0, stream>>>(Ybuf, z, feat_embed, b_ao, g1, be1, t1f, t1h);
    // FFN up: relu(t1 @ W1.T + b1) -> bf16   (N=1024)
    gemm_bf16_kernel<1><<<dim3(4 * DD / 64, NROW / 128), 256, 0, stream>>>(
        t1h, W1H, b1, nullptr, U, NROW, 4 * DD, DD);
    // FFN down: U @ W2.T -> f32 (b2 folded into LN2)   (K=1024)
    gemm_bf16_kernel<0><<<dim3(DD / 64, NROW / 128), 256, 0, stream>>>(
        U, W2H, nullptr, Ybuf, nullptr, NROW, DD, 4 * DD);
    // LN2 + Wop projection -> delta
    ln2_delta_kernel<<<NROW / 8, 256, 0, stream>>>(t1f, Ybuf, b2, g2, be2, Wop, bop, delta);
    // gate + z_out + output projection
    final_kernel<<<NSEQ, 64, 0, stream>>>(z, delta, Wg, bg, Amat, cvec, out_o, out_z);
}
